// CausalFFNN_90580860272876
// MI455X (gfx1250) — matrix operand order repaired and checked by reading, not compiled, not run on hardware
//
#include <hip/hip_runtime.h>
#include <math.h>
#include <stdint.h>

// ---------- types ----------
typedef __bf16 bf16;
typedef __attribute__((ext_vector_type(16))) __bf16 v16bf;
typedef __attribute__((ext_vector_type(8)))  float  v8f;
typedef __attribute__((ext_vector_type(4)))  unsigned int u32x4;
typedef __attribute__((ext_vector_type(4)))  float  f32x4;

#define N_ROWS 1024
#define D_IN   512
#define HID    256

// ---------- CDNA5 async global->LDS copy (ASYNCcnt-tracked, bypasses VGPRs) ----------
// LDS byte offset = low 32 bits of the generic pointer (addrspacecast(3->0) keeps
// the LDS offset in addr[31:0] per the flat-aperture rules).
__device__ inline void async_copy_b128(void* lds_dst, const void* gsrc) {
    unsigned lds_off = (unsigned)(uintptr_t)lds_dst;
    asm volatile("global_load_async_to_lds_b128 %0, %1, off"
                 :: "v"(lds_off), "v"(gsrc) : "memory");
}
__device__ inline void async_wait0() {
    asm volatile("s_wait_asynccnt 0x0" ::: "memory");
}

// ---------- WMMA wrapper (codegen-confirmed builtin, gfx1250) ----------
__device__ inline v8f wmma_bf16(v16bf a, v16bf b, v8f c) {
    // D(16x16,f32) = A(16x32,bf16) * B(32x16,bf16) + C
    return __builtin_amdgcn_wmma_f32_16x16x32_bf16(
        /*neg_a=*/false, a, /*neg_b=*/false, b,
        /*c_mod=*/(short)0, c, /*reuse_a=*/false, /*reuse_b=*/false);
}

// A fragment, row-major A[m][k], lda in elements.
// ISA layout (16-bit A 16x32): lane<16 -> M=lane, v0..3 = K 0..7, v4..7 = K 16..23
//                              lane>=16 -> M=lane-16, v0..3 = K 8..15, v4..7 = K 24..31
__device__ inline v16bf load_frag_a(const bf16* A, int lda, int m, int k0, int half) {
    union { v16bf v; u32x4 q[2]; } f;
    const bf16* p = A + (size_t)m * lda + k0 + 8 * half;
    f.q[0] = *(const u32x4*)(p);
    f.q[1] = *(const u32x4*)(p + 16);
    return f.v;
}

// B fragment from TRANSPOSED weights Bt[n][k] (row n = output column), ldb in elements.
// ISA layout (16-bit B 32x16): lane<16 -> N=lane, v0..3 = K 0..7, v4..7 = K 16..23
//                              lane>=16 -> N=lane-16, v0..3 = K 8..15, v4..7 = K 24..31
__device__ inline v16bf load_frag_b(const bf16* Bt, int ldb, int n, int k0, int half) {
    union { v16bf v; u32x4 q[2]; } f;
    const bf16* p = Bt + (size_t)n * ldb + k0 + 8 * half;
    f.q[0] = *(const u32x4*)(p);
    f.q[1] = *(const u32x4*)(p + 16);
    return f.v;
}

// ---------- conversion kernels ----------
__global__ void k_cvt_bf16(const float* __restrict__ src, bf16* __restrict__ dst, int n) {
    int i = blockIdx.x * blockDim.x + threadIdx.x;
    if (i < n) dst[i] = (bf16)src[i];
}

// W[K][N] (row-major) -> Wt[N][K] bf16
__global__ void k_cvtT_bf16(const float* __restrict__ W, bf16* __restrict__ Wt, int K, int N) {
    int i = blockIdx.x * blockDim.x + threadIdx.x;
    if (i < K * N) {
        int k = i / N, n = i % N;
        Wt[(size_t)n * K + k] = (bf16)W[i];
    }
}

// ---------- fused MLP chain: 16 rows per block, 16 waves (one N-tile each) ----------
__global__ __launch_bounds__(512)
void k_fused_mlp(const bf16* __restrict__ Ebf,   // [1024][512]
                 const bf16* __restrict__ W1t,   // [256][512]
                 const float* __restrict__ b1,
                 const bf16* __restrict__ W2t,   // [256][256]
                 const float* __restrict__ b2,
                 const bf16* __restrict__ Wat,   // [256][256]
                 const bf16* __restrict__ Wbt,   // [256][256]
                 const float* __restrict__ bp1,
                 float* __restrict__ Pi,         // [1024][256]
                 float* __restrict__ Pjb)        // [1024][256]  (Pj + bp1)
{
    __shared__ bf16 sE [16][D_IN];   // 16 KB: staged E block, shared by all 16 waves
    __shared__ bf16 sH1[16][HID];    //  8 KB
    __shared__ bf16 sHn[16][HID];    //  8 KB

    const int tid  = threadIdx.x;
    const int lane = tid & 31;
    const int wave = tid >> 5;            // N-tile index 0..15
    const int half = lane >> 4;
    const int mr   = lane & 15;           // A-row / B-col within tile
    const int col  = wave * 16 + mr;      // global output column for D frag
    const int rowBase = blockIdx.x * 16;

    // ---- async-stage the 16x512 bf16 E tile into LDS (1024 x 16B chunks) ----
    const bf16* Eblk = Ebf + (size_t)rowBase * D_IN;
    #pragma unroll
    for (int c = tid; c < 1024; c += 512) {
        int row  = c >> 6;            // 64 chunks per 1024-B row
        int col8 = (c & 63) << 3;     // 8 bf16 per 16-B chunk
        async_copy_b128(&sE[row][col8], Eblk + (size_t)row * D_IN + col8);
    }
    async_wait0();
    __syncthreads();

    const v8f vzero = {0.f, 0.f, 0.f, 0.f, 0.f, 0.f, 0.f, 0.f};

    // ---- layer 1: H1 = relu(E @ W1 + b1), K = 512 (A from LDS, B from global) ----
    v8f acc = vzero;
    const bf16* sEp = &sE[0][0];
    for (int k0 = 0; k0 < D_IN; k0 += 32) {
        v16bf a = load_frag_a(sEp, D_IN, mr, k0, half);
        v16bf b = load_frag_b(W1t, D_IN, col, k0, half);
        acc = wmma_bf16(a, b, acc);
    }
    {
        float bias = b1[col];
        #pragma unroll
        for (int r = 0; r < 8; ++r) {
            int row = r + 8 * half;       // D layout: v[r] -> M = r (+8 for upper lanes)
            float v = acc[r] + bias;
            sH1[row][col] = (bf16)(v > 0.f ? v : 0.f);
        }
    }
    __syncthreads();

    // ---- layer 2: Hn = relu(H1 @ W2 + b2), K = 256 ----
    acc = vzero;
    const bf16* sH1p = &sH1[0][0];
    for (int k0 = 0; k0 < HID; k0 += 32) {
        v16bf a = load_frag_a(sH1p, HID, mr, k0, half);
        v16bf b = load_frag_b(W2t, HID, col, k0, half);
        acc = wmma_bf16(a, b, acc);
    }
    {
        float bias = b2[col];
        #pragma unroll
        for (int r = 0; r < 8; ++r) {
            int row = r + 8 * half;
            float v = acc[r] + bias;
            sHn[row][col] = (bf16)(v > 0.f ? v : 0.f);
        }
    }
    __syncthreads();

    // ---- layers 3&4: Pi = Hn @ Wa ; Pjb = Hn @ Wb + bp1  (shared A fragments) ----
    v8f accA = vzero, accB = vzero;
    const bf16* sHnp = &sHn[0][0];
    for (int k0 = 0; k0 < HID; k0 += 32) {
        v16bf a  = load_frag_a(sHnp, HID, mr, k0, half);
        v16bf ba = load_frag_b(Wat, HID, col, k0, half);
        v16bf bb = load_frag_b(Wbt, HID, col, k0, half);
        accA = wmma_bf16(a, ba, accA);
        accB = wmma_bf16(a, bb, accB);
    }
    {
        float biasp = bp1[col];
        #pragma unroll
        for (int r = 0; r < 8; ++r) {
            int row = rowBase + r + 8 * half;
            Pi [(size_t)row * HID + col] = accA[r];
            Pjb[(size_t)row * HID + col] = accB[r] + biasp;
        }
    }
}

// ---------- pairwise: out[i][j] = softplus(sum_k relu(Pi[i,k]+Pjb[j,k]) * Wp2[k] + bp2) ----------
#define TPAD 260   // row stride: 260 % 64 == 4 -> conflict-free strided float4 access

__device__ inline float softplus_f(float x) {
    return fmaxf(x, 0.f) + log1pf(expf(-fabsf(x)));
}

// 64x64 (i,j) tile per 256-thread block; 4x4 register micro-tile per thread.
// DS:VALU per 4-k step = 9 x b128 loads : 192 VALU -> compute-dense inner loop.
__global__ __launch_bounds__(256)
void k_pairwise(const float* __restrict__ Pi, const float* __restrict__ Pjb,
                const float* __restrict__ Wp2, const float* __restrict__ bp2,
                float* __restrict__ out)
{
    __shared__ float sPi[64][TPAD];   // ~66.5 KB
    __shared__ float sPj[64][TPAD];   // ~66.5 KB
    __shared__ float sW[HID];

    const int t = threadIdx.x;
    const int iBase = blockIdx.y * 64;
    const int jBase = blockIdx.x * 64;

    // ---- async-stage both 64x256 f32 tiles (64 KB each; 4096 x 16B chunks per tile) ----
    #pragma unroll
    for (int c = t; c < 64 * 64; c += 256) {
        int row  = c >> 6;
        int col4 = (c & 63) << 2;     // 4 floats per 16-B chunk
        async_copy_b128(&sPi[row][col4], &Pi [(size_t)(iBase + row) * HID + col4]);
        async_copy_b128(&sPj[row][col4], &Pjb[(size_t)(jBase + row) * HID + col4]);
    }
    sW[t] = Wp2[t];
    async_wait0();
    __syncthreads();

    const int tx = t & 15;   // j within tile (+16b)
    const int ty = t >> 4;   // i within tile (+16a)

    float acc[4][4] = {};
    for (int k = 0; k < HID; k += 4) {
        f32x4 pi[4], pj[4];
        #pragma unroll
        for (int a = 0; a < 4; ++a) {
            pi[a] = *(const f32x4*)&sPi[ty + 16 * a][k];
            pj[a] = *(const f32x4*)&sPj[tx + 16 * a][k];
        }
        f32x4 wv = *(const f32x4*)&sW[k];
        #pragma unroll
        for (int u = 0; u < 4; ++u) {
            float w = wv[u];
            #pragma unroll
            for (int a = 0; a < 4; ++a)
                #pragma unroll
                for (int b = 0; b < 4; ++b)
                    acc[a][b] = fmaf(fmaxf(pi[a][u] + pj[b][u], 0.f), w, acc[a][b]);
        }
    }

    const float bb = bp2[0];
    #pragma unroll
    for (int a = 0; a < 4; ++a) {
        int i = iBase + ty + 16 * a;
        #pragma unroll
        for (int b = 0; b < 4; ++b) {
            int j = jBase + tx + 16 * b;
            out[(size_t)i * N_ROWS + j] = (i == j) ? 0.f : softplus_f(acc[a][b] + bb);
        }
    }
}

// ---------- launch ----------
extern "C" void kernel_launch(void* const* d_in, const int* in_sizes, int n_in,
                              void* d_out, int out_size, void* d_ws, size_t ws_size,
                              hipStream_t stream) {
    const float* E   = (const float*)d_in[0];   // 1024x512
    const float* W1  = (const float*)d_in[1];   // 512x256
    const float* b1  = (const float*)d_in[2];   // 256
    const float* W2  = (const float*)d_in[3];   // 256x256
    const float* b2  = (const float*)d_in[4];   // 256
    const float* Wp1 = (const float*)d_in[5];   // 512x256 (Wa=rows 0..255, Wb=rows 256..511)
    const float* bp1 = (const float*)d_in[6];   // 256
    const float* Wp2 = (const float*)d_in[7];   // 256x1
    const float* bp2 = (const float*)d_in[8];   // 1
    float* out = (float*)d_out;

    char* ws = (char*)d_ws;
    bf16* Ebf = (bf16*)ws;  ws += (size_t)N_ROWS * D_IN * sizeof(bf16);   // 1 MB
    bf16* W1t = (bf16*)ws;  ws += (size_t)HID * D_IN * sizeof(bf16);      // 256 KB
    bf16* W2t = (bf16*)ws;  ws += (size_t)HID * HID * sizeof(bf16);       // 128 KB
    bf16* Wat = (bf16*)ws;  ws += (size_t)HID * HID * sizeof(bf16);       // 128 KB
    bf16* Wbt = (bf16*)ws;  ws += (size_t)HID * HID * sizeof(bf16);       // 128 KB
    float* Pi  = (float*)ws; ws += (size_t)N_ROWS * HID * sizeof(float);  // 1 MB
    float* Pjb = (float*)ws;                                              // 1 MB

    // fp32 -> bf16 (weights transposed so WMMA fragments are contiguous 16B loads)
    k_cvt_bf16 <<<(N_ROWS * D_IN + 255) / 256, 256, 0, stream>>>(E, Ebf, N_ROWS * D_IN);
    k_cvtT_bf16<<<(D_IN * HID  + 255) / 256, 256, 0, stream>>>(W1, W1t, D_IN, HID);
    k_cvtT_bf16<<<(HID * HID   + 255) / 256, 256, 0, stream>>>(W2, W2t, HID, HID);
    k_cvtT_bf16<<<(HID * HID   + 255) / 256, 256, 0, stream>>>(Wp1,             Wat, HID, HID);
    k_cvtT_bf16<<<(HID * HID   + 255) / 256, 256, 0, stream>>>(Wp1 + HID * HID, Wbt, HID, HID);

    // fused WMMA MLP chain -> Pi, Pjb
    k_fused_mlp<<<N_ROWS / 16, 512, 0, stream>>>(Ebf, W1t, b1, W2t, b2, Wat, Wbt, bp1, Pi, Pjb);

    // pairwise softplus stage (64x64 tiles)
    k_pairwise<<<dim3(N_ROWS / 64, N_ROWS / 64), 256, 0, stream>>>(Pi, Pjb, Wp2, bp2, out);
}
